// BoltzmannMachine_82437602279699
// MI455X (gfx1250) — matrix operand order repaired and checked by reading, not compiled, not run on hardware
//
#include <hip/hip_runtime.h>
#include <hip/hip_bf16.h>
#include <cstdint>

// ---------------------------------------------------------------------------
// Types
// ---------------------------------------------------------------------------
typedef __attribute__((ext_vector_type(16))) _Float16 v16h;
typedef __attribute__((ext_vector_type(8)))  _Float16 v8h;
typedef __attribute__((ext_vector_type(4)))  _Float16 h4;
typedef __attribute__((ext_vector_type(8)))  float    v8f;
typedef __attribute__((ext_vector_type(4)))  unsigned int u32x4;
typedef __attribute__((ext_vector_type(8)))  int          i32x8;
typedef __attribute__((ext_vector_type(4)))  int          i32x4;

#define N_UNITS 8192
#define BLK     128              // Gibbs block (perm order)
#define NBLK    (N_UNITS / BLK)  // 64
#define KSPLIT  4
#define KRANGE  (N_UNITS / KSPLIT) // 2048
#define KCH     128              // K-chunk (halves) per LDS tile
#define NCH     (KRANGE / KCH)   // 16

// ws layout (bytes). Requires ~128.3 MB of scratch (W in f16 + vectors).
#define WF16_BYTES ((size_t)N_UNITS * (size_t)N_UNITS * 2)
#define S_OFF      (WF16_BYTES)
#define SPART_OFF  (S_OFF     + (size_t)N_UNITS * 4)
#define THR_OFF    (SPART_OFF + (size_t)KSPLIT * N_UNITS * 4)
#define ST_OFF     (THR_OFF   + (size_t)N_UNITS * 4)
#define STH_OFF    (ST_OFF    + (size_t)N_UNITS * 4)
#define DL_OFF     (STH_OFF   + (size_t)N_UNITS * 2)
#define FLG_OFF    (DL_OFF    + (size_t)N_UNITS * 4)

// TDM only in the device pass; host pass always takes the fallback body.
#if defined(__HIP_DEVICE_COMPILE__) && defined(__gfx1250__)
#if __has_builtin(__builtin_amdgcn_tensor_load_to_lds) && __has_builtin(__builtin_amdgcn_s_wait_tensorcnt)
#define HAVE_TDM 1
#endif
#endif

// ---------------------------------------------------------------------------
// K1: W f32 -> f16 (one pass; afterwards W_f16 (128MB) is L2-resident)
// ---------------------------------------------------------------------------
__global__ void __launch_bounds__(256) k_convert(const float* __restrict__ w,
                                                 _Float16* __restrict__ wf) {
  size_t i = ((size_t)blockIdx.x * blockDim.x + threadIdx.x) * 4;
  float4 v = *(const float4*)(w + i);
  h4 h;
  h[0] = (_Float16)v.x; h[1] = (_Float16)v.y;
  h[2] = (_Float16)v.z; h[3] = (_Float16)v.w;
  *(h4*)(wf + i) = h;
}

// ---------------------------------------------------------------------------
// K2: thresholds thr = T*logit(u)  (u <= sigmoid(x/T)  <=>  x >= thr),
//     state copies, flag reset (re-run every call => deterministic replays)
// ---------------------------------------------------------------------------
__global__ void __launch_bounds__(256) k_prep(const float* __restrict__ state_in,
                                              const float* __restrict__ uniforms,
                                              const float* __restrict__ T,
                                              float* __restrict__ thr,
                                              float* __restrict__ st,
                                              _Float16* __restrict__ sth,
                                              int* __restrict__ flags) {
  int j = blockIdx.x * blockDim.x + threadIdx.x;
  if (j < N_UNITS) {
    float u  = uniforms[j];
    float Tv = T[0];
    thr[j] = Tv * logf(u / (1.0f - u));
    float s = state_in[j];
    st[j]  = s;
    sth[j] = (_Float16)s;
  }
  if (j == 0) { flags[0] = 0; flags[1] = 0; }
}

// ---------------------------------------------------------------------------
// TDM: async 2D tile (BLK rows x KCH halves) of W_f16 -> LDS
// ---------------------------------------------------------------------------
#if defined(HAVE_TDM)
__device__ __forceinline__ void tdm_load_tile(unsigned lds_off, const _Float16* gptr) {
  uint64_t ga = (uint64_t)(uintptr_t)gptr;
  u32x4 g0;
  g0[0] = 1u;                                   // count=1, user descriptor
  g0[1] = lds_off;                              // LDS byte address
  g0[2] = (unsigned)(ga & 0xffffffffu);         // global_addr[31:0]
  g0[3] = (unsigned)((ga >> 32) & 0x1ffffffu) | (2u << 30); // [56:32] | type=2
  i32x8 g1;
  g1[0] = (1 << 16);                            // data_size = 2 bytes
  g1[1] = (int)((N_UNITS & 0xffff) << 16);      // tensor_dim0 lo16
  g1[2] = (int)((N_UNITS >> 16) | ((N_UNITS & 0xffff) << 16)); // dim0 hi | dim1 lo
  g1[3] = (int)((N_UNITS >> 16) | (KCH << 16)); // dim1 hi | tile_dim0
  g1[4] = BLK;                                  // tile_dim1 (tile_dim2 = 0)
  g1[5] = N_UNITS;                              // tensor_dim0_stride lo32
  g1[6] = 0;
  g1[7] = 0;
  i32x4 z4 = (i32x4)0;
#if defined(__clang_major__) && __clang_major__ >= 23
  i32x8 z8 = (i32x8)0;
  __builtin_amdgcn_tensor_load_to_lds(g0, g1, z4, z4, z8, 0);
#else
  __builtin_amdgcn_tensor_load_to_lds(g0, g1, z4, z4, 0);
#endif
}
#endif

// ---------------------------------------------------------------------------
// K3: S0 = W_f16 @ state0 via WMMA f16 (state broadcast into all 16 B cols).
// grid = (64 row-blocks, 4 K-splits); deterministic split-K via Spart.
// ---------------------------------------------------------------------------
__global__ void __launch_bounds__(256) k_gemv(const _Float16* __restrict__ wf,
                                              const _Float16* __restrict__ sth,
                                              float* __restrict__ spart) {
  __shared__ _Float16 tile[2][BLK * KCH];   // 2 x 32KB, TDM double buffer
  const int br = blockIdx.x, ks = blockIdx.y;
  const int tid = threadIdx.x, lane = tid & 31, wave = tid >> 5;
  const int kbase0 = ks * KRANGE;
  const _Float16* wbase = wf + (size_t)br * BLK * N_UNITS;
  v8f acc = {};

#if defined(HAVE_TDM)
  if (wave == 0)
    tdm_load_tile((unsigned)(uintptr_t)(void*)&tile[0][0], wbase + kbase0);
#endif

  for (int c = 0; c < NCH; ++c) {
#if defined(HAVE_TDM)
    if (wave == 0) {
      if (c + 1 < NCH) {
        tdm_load_tile((unsigned)(uintptr_t)(void*)&tile[(c + 1) & 1][0],
                      wbase + kbase0 + (size_t)(c + 1) * KCH);
        __builtin_amdgcn_s_wait_tensorcnt(1);  // chunk c done, c+1 in flight
      } else {
        __builtin_amdgcn_s_wait_tensorcnt(0);  // last chunk: drain
      }
    }
    __syncthreads();
    const _Float16* tl = &tile[c & 1][0];
#else
    // Fallback: synchronous cooperative copy, single buffer.
    for (int e = tid; e < (BLK * KCH) / 8; e += 256) {
      int r = e / (KCH / 8), c8 = e % (KCH / 8);
      *(uint4*)(&tile[0][0] + r * KCH + c8 * 8) =
          *(const uint4*)(wbase + (size_t)r * N_UNITS + kbase0 + c * KCH + c8 * 8);
    }
    __syncthreads();
    const _Float16* tl = &tile[0][0];
#endif
    const int m = lane & 15;
    const int khalf = (lane >> 4) * 8;
    const int kg = kbase0 + c * KCH;
#pragma unroll
    for (int kb = 0; kb < KCH; kb += 32) {
      // A (16x32 f16): lanes 0-15 hold K=[kb+0..7 | kb+16..23], 16-31 offset +8
      v8h alo = *(const v8h*)(tl + (wave * 16 + m) * KCH + kb + khalf);
      v8h ahi = *(const v8h*)(tl + (wave * 16 + m) * KCH + kb + 16 + khalf);
      v16h a = __builtin_shufflevector(alo, ahi, 0, 1, 2, 3, 4, 5, 6, 7,
                                       8, 9, 10, 11, 12, 13, 14, 15);
      // B (32x16): B[k][n] = s[k] for all n -> lanes 0-15: K=[kb+0..7 | kb+16..23],
      // lanes 16-31 offset +8 (same K order as A)
      v8h blo = *(const v8h*)(sth + kg + kb + khalf);
      v8h bhi = *(const v8h*)(sth + kg + kb + 16 + khalf);
      v16h b = __builtin_shufflevector(blo, bhi, 0, 1, 2, 3, 4, 5, 6, 7,
                                       8, 9, 10, 11, 12, 13, 14, 15);
      acc = __builtin_amdgcn_wmma_f32_16x16x32_f16(false, a, false, b,
                                                   (short)0, acc, false, false);
    }
    __syncthreads();
  }
  // Column N=0 lives in lanes 0 (M=0..7) and 16 (M=8..15).
  if ((lane & 15) == 0) {
    int rowb = br * BLK + wave * 16 + (lane >> 4) * 8;
#pragma unroll
    for (int r = 0; r < 8; ++r)
      spart[(size_t)ks * N_UNITS + rowb + r] = acc[r];
  }
}

__global__ void __launch_bounds__(256) k_reduce(const float* __restrict__ spart,
                                                float* __restrict__ S) {
  int j = blockIdx.x * blockDim.x + threadIdx.x;
  S[j] = spart[j] + spart[N_UNITS + j] + spart[2 * N_UNITS + j] +
         spart[3 * N_UNITS + j];
}

// ---------------------------------------------------------------------------
// K4: persistent sweep. WG0: per-block serial chain (wave 0, readlane
// broadcasts, 4 rounds of 32). WGs 1..63: fold flips into S (L2-resident
// f16 rows), agent-scope acquire/release handshake.
// ---------------------------------------------------------------------------
__global__ void __launch_bounds__(256) k_sweep(const _Float16* __restrict__ wf,
                                               const int* __restrict__ perm,
                                               const float* __restrict__ clamping,
                                               float* __restrict__ S,
                                               const float* __restrict__ thr,
                                               float* __restrict__ st,
                                               float* __restrict__ dl,
                                               int* __restrict__ flags) {
  const int tid = threadIdx.x;
  if (blockIdx.x == 0) {
    __shared__ int   u[BLK];
    __shared__ float sval[BLK], thrv[BLK], oldv[BLK], freev[BLK], dlt[BLK];
    __shared__ _Float16 cc[BLK * BLK];  // 32KB in-block coupling
    for (int b = 0; b < NBLK; ++b) {
      if (tid == 0) {  // wait: S holds all flips from blocks < b
        while (__hip_atomic_load(&flags[1], __ATOMIC_ACQUIRE,
                                 __HIP_MEMORY_SCOPE_AGENT) < 63 * b)
          __builtin_amdgcn_s_sleep(2);
      }
      __syncthreads();
      __threadfence();
      if (tid < BLK) {
        int uu = perm[b * BLK + tid];
        u[tid]     = uu;
        sval[tid]  = S[uu];
        thrv[tid]  = thr[uu];
        oldv[tid]  = st[uu];
        freev[tid] = (clamping[uu] == 0.0f) ? 1.0f : 0.0f;
      }
      __syncthreads();
      // coupling c[j][m] = w[u_j][u_m] (== w[u_m][u_j], W symmetric); L2 hits
      for (int e = tid; e < BLK * BLK; e += 256) {
        int j = e >> 7, m2 = e & 127;
        cc[e] = wf[(size_t)u[m2] * N_UNITS + u[j]];
      }
      __syncthreads();
      if (tid < 32) {  // serial chain on wave 0
        const int lane = tid;
        for (int r = 0; r < 4; ++r) {
          int j = r * 32 + lane;
          float acc = sval[j], th = thrv[j], od = oldv[j], fr = freev[j];
          for (int m2 = 0; m2 < r * 32; ++m2)  // earlier rounds' deltas
            acc += (float)cc[j * BLK + m2] * dlt[m2];
          float cm[32];
#pragma unroll
          for (int m2 = 0; m2 < 32; ++m2)
            cm[m2] = (float)cc[j * BLK + r * 32 + m2];
          float my = 0.0f;
#pragma unroll
          for (int m2 = 0; m2 < 32; ++m2) {
            float dec  = (acc >= th) ? 1.0f : -1.0f;
            float cand = fr * (dec - od);                 // clamped -> 0
            float dm   = __int_as_float(
                __builtin_amdgcn_readlane(__float_as_int(cand), m2));
            if (lane == m2) my = cand;                    // capture own delta
            acc += cm[m2] * dm;
          }
          dlt[j] = my;
        }
      }
      __syncthreads();
      if (tid < BLK) {
        float d = dlt[tid];
        st[u[tid]]        = oldv[tid] + d;
        dl[b * BLK + tid] = d;
      }
      __threadfence();
      __syncthreads();
      if (tid == 0)
        __hip_atomic_store(&flags[0], b + 1, __ATOMIC_RELEASE,
                           __HIP_MEMORY_SCOPE_AGENT);
    }
  } else {
    const int w = blockIdx.x - 1;            // 0..62
    const int chunk = (N_UNITS + 62) / 63;   // 131 columns per WG
    const int i0 = w * chunk;
    const int i1 = (i0 + chunk < N_UNITS) ? (i0 + chunk) : N_UNITS;
    for (int b = 0; b < NBLK; ++b) {
      if (tid == 0) {
        while (__hip_atomic_load(&flags[0], __ATOMIC_ACQUIRE,
                                 __HIP_MEMORY_SCOPE_AGENT) < b + 1)
          __builtin_amdgcn_s_sleep(2);
      }
      __syncthreads();
      __threadfence();
      for (int m2 = 0; m2 < BLK; ++m2) {
        float d = dl[b * BLK + m2];
        if (d != 0.0f) {
          int uu = perm[b * BLK + m2];
          const _Float16* row = wf + (size_t)uu * N_UNITS;
          for (int i = i0 + tid; i < i1; i += 256)
            S[i] += d * (float)row[i];     // exclusive column ranges: no races
        }
      }
      __threadfence();
      __syncthreads();
      if (tid == 0)
        __hip_atomic_fetch_add(&flags[1], 1, __ATOMIC_ACQ_REL,
                               __HIP_MEMORY_SCOPE_AGENT);
    }
  }
}

__global__ void __launch_bounds__(256) k_finish(const float* __restrict__ st,
                                                float* __restrict__ out) {
  int j = blockIdx.x * blockDim.x + threadIdx.x;
  out[j] = st[j];
}

// ---------------------------------------------------------------------------
// Launch
// ---------------------------------------------------------------------------
extern "C" void kernel_launch(void* const* d_in, const int* in_sizes, int n_in,
                              void* d_out, int out_size, void* d_ws, size_t ws_size,
                              hipStream_t stream) {
  (void)in_sizes; (void)n_in; (void)out_size; (void)ws_size;
  const float* w        = (const float*)d_in[0];
  const float* state0   = (const float*)d_in[1];
  const float* clamping = (const float*)d_in[2];
  const float* T        = (const float*)d_in[3];
  const int*   perm     = (const int*)d_in[4];
  const float* uniforms = (const float*)d_in[5];
  float* out = (float*)d_out;

  char* ws = (char*)d_ws;  // needs >= FLG_OFF + 64 bytes (~128.3 MB)
  _Float16* wf    = (_Float16*)(ws);
  float*    S     = (float*)(ws + S_OFF);
  float*    spart = (float*)(ws + SPART_OFF);
  float*    thr   = (float*)(ws + THR_OFF);
  float*    st    = (float*)(ws + ST_OFF);
  _Float16* sth   = (_Float16*)(ws + STH_OFF);
  float*    dl    = (float*)(ws + DL_OFF);
  int*      flags = (int*)(ws + FLG_OFF);

  k_convert<<<(N_UNITS * (size_t)N_UNITS) / (256 * 4), 256, 0, stream>>>(w, wf);
  k_prep<<<N_UNITS / 256, 256, 0, stream>>>(state0, uniforms, T, thr, st, sth, flags);
  k_gemv<<<dim3(NBLK, KSPLIT), 256, 0, stream>>>(wf, sth, spart);
  k_reduce<<<N_UNITS / 256, 256, 0, stream>>>(spart, S);
  k_sweep<<<64, 256, 0, stream>>>(wf, perm, clamping, S, thr, st, dl, flags);
  k_finish<<<N_UNITS / 256, 256, 0, stream>>>(st, out);
}
